// GraphCNN_69363721831139
// MI455X (gfx1250) — matrix operand order repaired and checked by reading, not compiled, not run on hardware
//
#include <hip/hip_runtime.h>

typedef __bf16 bf16_t;
typedef __attribute__((ext_vector_type(16))) __bf16 v16bf;
typedef __attribute__((ext_vector_type(8)))  __bf16 v8bf;
typedef __attribute__((ext_vector_type(8)))  float  v8f;

#define DD 128          // hidden dim == GEMM N-width, fixed
#define LDS_STRIDE 40   // 32 K-values + 8 pad (bf16 units) -> 80B rows, 16B aligned

__device__ __forceinline__ bf16_t f2bf(float f) {
  union { float f; unsigned u; } x; x.f = f;
  unsigned r = x.u + 0x7FFFu + ((x.u >> 16) & 1u);   // round-to-nearest-even
  unsigned short h = (unsigned short)(r >> 16);
  return __builtin_bit_cast(bf16_t, h);
}

// C[M,128] = A[M,K] * B[K,128] (+bias).
// One wave -> 32x128 strip: 2 A-fragments x 8 B-tiles = 16 WMMAs per K-step,
// each LDS B-fragment feeds 2 WMMAs. 256-thread block (8 waves) -> 256x128.
// B tile [32x128] staged transposed into double-buffered LDS; next tile is
// prefetched into registers while the current tile's WMMAs run.
template<bool A_IS_F32>
__global__ __launch_bounds__(256)
void gemm_wmma(const void* __restrict__ Ap, const bf16_t* __restrict__ Bp,
               float* __restrict__ outF, bf16_t* __restrict__ outB,
               const float* __restrict__ bias,
               int lda, int K, long strideA, long strideB, long strideC,
               int blocksPerBatch)
{
  __shared__ bf16_t lds[2][DD * LDS_STRIDE];
  const int tid   = threadIdx.x;
  const int wave  = tid >> 5;          // wave32
  const int lane  = tid & 31;
  const int batch = blockIdx.x / blocksPerBatch;
  const int mblk  = blockIdx.x % blocksPerBatch;
  const int mbase = mblk * 256 + wave * 32;   // wave owns rows [mbase, mbase+32)
  const int mrow  = lane & 15;         // A row in tile == B/C column lane
  const int khalf = lane >> 4;

  const bf16_t* Bb = Bp + (long)batch * strideB;

  const int kr = tid >> 4;             // 0..15 : B tile row pair base
  const int nc = (tid & 15) * 8;       // col base for staging

  auto load_rows = [&](int kb, v8bf* r) {
    #pragma unroll
    for (int rr = 0; rr < 2; ++rr)
      r[rr] = *(const v8bf*)(Bb + (long)(kb + kr + rr * 16) * DD + nc);
  };
  auto store_rows = [&](int buf, const v8bf* r) {
    #pragma unroll
    for (int rr = 0; rr < 2; ++rr) {
      const int k = kr + rr * 16;
      #pragma unroll
      for (int j = 0; j < 8; ++j)
        lds[buf][(nc + j) * LDS_STRIDE + k] = r[rr][j];
    }
  };

  v8f acc[2][8] = {};

  // prologue: stage first B tile
  {
    v8bf s0[2];
    load_rows(0, s0);
    store_rows(0, s0);
  }
  __syncthreads();

  int cur = 0;
  for (int kb = 0; kb < K; kb += 32) {
    const bool has_next = (kb + 32 < K);
    v8bf nxt[2];
    if (has_next) load_rows(kb + 32, nxt);   // long-latency global loads issue early

    // ---- 2 A fragments: lane needs K = khalf*8+(0..7) and 16+khalf*8+(0..7) ----
    v16bf afrag[2];
    #pragma unroll
    for (int mt = 0; mt < 2; ++mt) {
      const long aoff = (long)batch * strideA +
                        (long)(mbase + mt * 16 + mrow) * lda + kb + khalf * 8;
      if (A_IS_F32) {
        const float* ap = (const float*)Ap + aoff;
        v8f a0 = *(const v8f*)(ap);
        v8f a1 = *(const v8f*)(ap + 16);
        #pragma unroll
        for (int j = 0; j < 8; ++j) { afrag[mt][j] = f2bf(a0[j]); afrag[mt][8 + j] = f2bf(a1[j]); }
      } else {
        const bf16_t* ap = (const bf16_t*)Ap + aoff;
        v8bf a0 = *(const v8bf*)(ap);
        v8bf a1 = *(const v8bf*)(ap + 16);
        #pragma unroll
        for (int j = 0; j < 8; ++j) { afrag[mt][j] = a0[j]; afrag[mt][8 + j] = a1[j]; }
      }
    }

    // ---- 8 B fragments; each feeds 2 WMMAs (one per A fragment) ----
    #pragma unroll
    for (int t = 0; t < 8; ++t) {
      const bf16_t* bl = &lds[cur][(t * 16 + mrow) * LDS_STRIDE + khalf * 8];
      v8bf b0 = *(const v8bf*)(bl);
      v8bf b1 = *(const v8bf*)(bl + 16);
      v16bf bfrag;
      #pragma unroll
      for (int j = 0; j < 8; ++j) { bfrag[j] = b0[j]; bfrag[8 + j] = b1[j]; }
      acc[0][t] = __builtin_amdgcn_wmma_f32_16x16x32_bf16(
          false, afrag[0], false, bfrag, (short)0, acc[0][t], false, false);
      acc[1][t] = __builtin_amdgcn_wmma_f32_16x16x32_bf16(
          false, afrag[1], false, bfrag, (short)0, acc[1][t], false, false);
    }

    if (has_next) store_rows(cur ^ 1, nxt);  // fill other buffer after our reads
    __syncthreads();
    cur ^= 1;
  }

  // ---- epilogue: C lane layout row = i + 8*khalf, col = lane&15 ----
  #pragma unroll
  for (int mt = 0; mt < 2; ++mt) {
    #pragma unroll
    for (int t = 0; t < 8; ++t) {
      const int n = t * 16 + mrow;
      const float bv = bias ? bias[n] : 0.f;
      #pragma unroll
      for (int i = 0; i < 8; ++i) {
        const long off = (long)batch * strideC +
                         (long)(mbase + mt * 16 + khalf * 8 + i) * DD + n;
        const float v = acc[mt][t][i] + bv;
        if (outF) outF[off] = v;
        if (outB) outB[off] = f2bf(v);
      }
    }
  }
}

__global__ void cvt_f32_to_bf16(const float* __restrict__ in, bf16_t* __restrict__ out, long n) {
  long i = (long)blockIdx.x * blockDim.x + threadIdx.x;
  const long stride = (long)gridDim.x * blockDim.x;
  for (; i < n; i += stride) out[i] = f2bf(in[i]);
}

__global__ void zero_f32(float* __restrict__ p, int n) {
  const int i = blockIdx.x * blockDim.x + threadIdx.x;
  if (i < n) p[i] = 0.f;
}

__global__ __launch_bounds__(128)
void bn_stats(const float* __restrict__ z, float* __restrict__ sum,
              float* __restrict__ sumsq, int rowsPerBlock) {
  const int col = threadIdx.x;
  const long r0 = (long)blockIdx.x * rowsPerBlock;
  float s = 0.f, s2 = 0.f;
  for (int r = 0; r < rowsPerBlock; ++r) {
    const float v = z[(r0 + r) * DD + col];
    s += v; s2 += v * v;
  }
  atomicAdd(&sum[col], s);
  atomicAdd(&sumsq[col], s2);
}

__global__ __launch_bounds__(256)
void bn_apply_relu(const float* __restrict__ z, const float* __restrict__ sum,
                   const float* __restrict__ sumsq, const float* __restrict__ gamma,
                   const float* __restrict__ beta, float* __restrict__ outF,
                   bf16_t* __restrict__ outB, int rows)
{
  const long n = (long)rows * DD;
  long i = (long)blockIdx.x * blockDim.x + threadIdx.x;
  const long stride = (long)gridDim.x * blockDim.x;
  const float inv_rows = 1.f / (float)rows;
  for (; i < n; i += stride) {
    const int col = (int)(i & (DD - 1));
    const float mean = sum[col] * inv_rows;
    const float var  = sumsq[col] * inv_rows - mean * mean;  // biased, matches ref
    const float sc   = gamma[col] * rsqrtf(var + 1e-5f);
    float v = (z[i] - mean) * sc + beta[col];
    v = v > 0.f ? v : 0.f;
    if (outF) outF[i] = v;
    if (outB) outB[i] = f2bf(v);
  }
}

extern "C" void kernel_launch(void* const* d_in, const int* in_sizes, int n_in,
                              void* d_out, int out_size, void* d_ws, size_t ws_size,
                              hipStream_t stream) {
  const float* x     = (const float*)d_in[0];
  // d_in[1] = padded_nei: unused by the reference computation
  const float* adj   = (const float*)d_in[2];
  const float* W1    = (const float*)d_in[3];
  const float* b1    = (const float*)d_in[4];
  const float* W2    = (const float*)d_in[5];
  const float* b2    = (const float*)d_in[6];
  const float* g_in  = (const float*)d_in[7];
  const float* be_in = (const float*)d_in[8];
  const float* g_out = (const float*)d_in[9];
  const float* be_out= (const float*)d_in[10];

  const int B = 32, N = 1024, L = 2;
  const long BND = (long)B * N * DD;      // 4,194,304 elements

  char* w = (char*)d_ws;
  auto carve = [&](size_t bytes) { char* p = w; w += (bytes + 255) & ~(size_t)255; return p; };
  bf16_t* h_bf    = (bf16_t*)carve(BND * 2);
  bf16_t* pool_bf = (bf16_t*)carve(BND * 2);
  bf16_t* z_bf    = (bf16_t*)carve(BND * 2);
  float*  z       = (float*) carve(BND * 4);
  bf16_t* W1bf    = (bf16_t*)carve((size_t)L * DD * DD * 2);
  bf16_t* W2bf    = (bf16_t*)carve((size_t)L * DD * DD * 2);
  float*  stat    = (float*) carve(256 * 4);
  float* ssum = stat;
  float* ssq  = stat + 128;

  cvt_f32_to_bf16<<<1024, 256, 0, stream>>>(x, h_bf, BND);
  cvt_f32_to_bf16<<<64, 256, 0, stream>>>(W1, W1bf, (long)L * DD * DD);
  cvt_f32_to_bf16<<<64, 256, 0, stream>>>(W2, W2bf, (long)L * DD * DD);

  const int ROWS = B * N;                  // 32768
  for (int l = 0; l < L; ++l) {
    // pooled = adj @ h   (A: f32 [B,N,N], B: bf16 [B,N,128]) -> bf16
    gemm_wmma<true><<<B * (N / 256), 256, 0, stream>>>(
        adj, h_bf, nullptr, pool_bf, nullptr,
        N, N, (long)N * N, (long)N * DD, (long)N * DD, N / 256);

    // z = pooled @ W1[l] + b1[l]  -> f32
    gemm_wmma<false><<<ROWS / 256, 256, 0, stream>>>(
        pool_bf, W1bf + (long)l * DD * DD, z, nullptr, b1 + l * DD,
        DD, DD, 0, 0, 0, ROWS / 256);

    zero_f32<<<1, 256, 0, stream>>>(stat, 256);
    bn_stats<<<ROWS / 256, 128, 0, stream>>>(z, ssum, ssq, 256);
    bn_apply_relu<<<2048, 256, 0, stream>>>(z, ssum, ssq, g_in + l * DD, be_in + l * DD,
                                            nullptr, z_bf, ROWS);

    // z = relu_bn(z) @ W2[l] + b2[l]  -> f32
    gemm_wmma<false><<<ROWS / 256, 256, 0, stream>>>(
        z_bf, W2bf + (long)l * DD * DD, z, nullptr, b2 + l * DD,
        DD, DD, 0, 0, 0, ROWS / 256);

    zero_f32<<<1, 256, 0, stream>>>(stat, 256);
    bn_stats<<<ROWS / 256, 128, 0, stream>>>(z, ssum, ssq, 256);
    float* hF = (l == L - 1) ? (float*)d_out : nullptr;
    bn_apply_relu<<<2048, 256, 0, stream>>>(z, ssum, ssq, g_out + l * DD, be_out + l * DD,
                                            hF, h_bf, ROWS);
  }
}
